// CellDetectorWithClassifierInd_74517682586190
// MI455X (gfx1250) — matrix operand order repaired and checked by reading, not compiled, not run on hardware
//
#include <hip/hip_runtime.h>
#include <hip/hip_bf16.h>

#define B_      4
#define H_      512
#define W_      512
#define KTOP    512
#define CAP_    32768
#define NMS_THR_ 0.2f

typedef __attribute__((ext_vector_type(16))) _Float16 v16h;
typedef __attribute__((ext_vector_type(8)))  _Float16 v8h;
typedef __attribute__((ext_vector_type(8)))  float    v8f;
typedef __attribute__((ext_vector_type(4)))  unsigned v4u;
typedef __attribute__((ext_vector_type(4)))  int      v4i;
typedef __attribute__((ext_vector_type(8)))  int      v8i;

#if defined(__has_builtin)
#  if __has_builtin(__builtin_amdgcn_tensor_load_to_lds)
#    define HAVE_TDM_LD 1
#  endif
#  if __has_builtin(__builtin_amdgcn_tensor_store_from_lds)
#    define HAVE_TDM_ST 1
#  endif
#endif
#ifndef HAVE_TDM_LD
#  define HAVE_TDM_LD 0
#endif
#ifndef HAVE_TDM_ST
#  define HAVE_TDM_ST 0
#endif

static __device__ __forceinline__ float leaky_(float x){ return x > 0.f ? x : 0.1f * x; }

static __device__ __forceinline__ v16h cat16(v8h lo, v8h hi){
  v16h r;
#pragma unroll
  for(int i=0;i<8;++i){ r[i]=lo[i]; r[i+8]=hi[i]; }
  return r;
}

static __device__ __forceinline__ v8f wmma_f16(v16h a, v16h b, v8f c){
  // D = A(16x32 f16) * B(32x16 f16) + C(16x16 f32)
  return __builtin_amdgcn_wmma_f32_16x16x32_f16(false, a, false, b, (short)0, c, false, false);
}

static __device__ __forceinline__ void wait_tensor0(){
#if defined(__has_builtin)
#  if __has_builtin(__builtin_amdgcn_s_wait_tensorcnt)
  __builtin_amdgcn_s_wait_tensorcnt(0);
#  else
  asm volatile("s_wait_tensorcnt 0x0" ::: "memory");
#  endif
#else
  asm volatile("s_wait_tensorcnt 0x0" ::: "memory");
#endif
}

// ---------------------------------------------------------------------------
// Tensor Data Mover: 1-D contiguous tile, data_size = 8B (ISA 8.3/8.4).
// D# group0: count=1 | lds_addr | global_addr[56:0] | type=2
// D# group1: data_size=3(8B); tensor_dim0 = tile_dim0 = n8; tile_dim1=0 (1-D)
// ---------------------------------------------------------------------------
static __device__ __forceinline__ void tdm_desc_1d(const void* lds_p, const void* g_p,
                                                   unsigned nbytes, v4u& g0, v8i& g1){
  unsigned lds_addr = (unsigned)(unsigned long long)lds_p;   // low 32b of flat LDS addr = LDS offset
  unsigned long long ga = (unsigned long long)g_p;
  unsigned n8 = nbytes >> 3;
  g0.x = 1u;                                                 // count=1, user descriptor
  g0.y = lds_addr;
  g0.z = (unsigned)(ga & 0xffffffffu);
  g0.w = (unsigned)((ga >> 32) & 0x1ffffffu) | (2u << 30);   // global_addr[56:32] | type=2
  g1[0] = (int)(3u << 16);                                   // workgroup_mask=0, data_size=8B
  g1[1] = (int)((n8 & 0xffffu) << 16);                       // tensor_dim0[15:0]
  g1[2] = (int)(((n8 >> 16) & 0xffffu) | (1u << 16));        // tensor_dim0[31:16], tensor_dim1=1
  g1[3] = (int)((n8 & 0xffffu) << 16);                       // tile_dim0 = n8
  g1[4] = 0;                                                 // tile_dim1=0 (1-D), tile_dim2=0
  g1[5] = (int)n8;                                           // tensor_dim0_stride
  g1[6] = 0;
  g1[7] = 0;
}

#if HAVE_TDM_LD
static __device__ __forceinline__ void tdm_load_1d(void* lds_dst, const void* gsrc, unsigned nbytes){
  v4u g0; v8i g1;
  tdm_desc_1d(lds_dst, gsrc, nbytes, g0, g1);
  v4i gz = {0,0,0,0};
#if __clang_major__ >= 23
  v8i gz8 = {0,0,0,0,0,0,0,0};
  __builtin_amdgcn_tensor_load_to_lds(g0, g1, gz, gz, gz8, 0);
#else
  __builtin_amdgcn_tensor_load_to_lds(g0, g1, gz, gz, 0);
#endif
}
#endif

#if HAVE_TDM_ST
static __device__ __forceinline__ void tdm_store_1d(void* gdst, const void* lds_src, unsigned nbytes){
  v4u g0; v8i g1;
  tdm_desc_1d(lds_src, gdst, nbytes, g0, g1);
  v4i gz = {0,0,0,0};
#if __clang_major__ >= 23
  v8i gz8 = {0,0,0,0,0,0,0,0};
  __builtin_amdgcn_tensor_store_from_lds(g0, g1, gz, gz, gz8, 0);
#else
  __builtin_amdgcn_tensor_store_from_lds(g0, g1, gz, gz, 0);
#endif
}
#endif

// Stage a contiguous f16 tile global -> LDS (TDM if available), then barrier.
static __device__ __forceinline__ void stage_in(_Float16* sdst, const _Float16* gsrc,
                                                int nhalf, int tid){
#if HAVE_TDM_LD
  if(tid < 32){                        // one wave issues the DMA
    tdm_load_1d(sdst, gsrc, (unsigned)nhalf * 2u);
    wait_tensor0();
  }
  __syncthreads();
#else
  for(int i = tid; i < nhalf; i += 256) sdst[i] = gsrc[i];
  __syncthreads();
#endif
}

// Drain a contiguous f16 tile LDS -> global. Caller guarantees a barrier
// between the LDS writes and this call.
static __device__ __forceinline__ void drain_out(_Float16* gdst, const _Float16* ssrc,
                                                 int nhalf, int tid){
#if HAVE_TDM_ST
  if(tid < 32){
    tdm_store_1d(gdst, ssrc, (unsigned)nhalf * 2u);
    wait_tensor0();
  }
#else
  for(int i = tid; i < nhalf; i += 256) gdst[i] = ssrc[i];
#endif
}

// ---------------------------------------------------------------------------
// Weight prep: pack f32 weights into B-fragment-native f16 layout:
//   dst[((kb*16 + nt)*32 + lane)*16 + j] = W[o = nt*16 + (lane&15)][k]
//   with k = kb*32 + ((lane>>4)<<3) + (j&7) + ((j>>3)<<4)   (ISA 16-bit B 32x16 striping)
// ---------------------------------------------------------------------------
__global__ void k_prep_dense(const float* __restrict__ w, _Float16* __restrict__ dst, int Ktot){
  int nkb = Ktot >> 5;
  int total = nkb * 16 * 512;
  for(int i = blockIdx.x*blockDim.x + threadIdx.x; i < total; i += gridDim.x*blockDim.x){
    int j = i & 15, lane = (i >> 4) & 31, nt = (i >> 9) & 15, kb = i >> 13;
    int k = kb*32 + ((lane >> 4) << 3) + (j & 7) + ((j >> 3) << 4);
    int o = nt*16 + (lane & 15);
    dst[i] = (_Float16)w[o*Ktot + k];
  }
}

// conv weights stored [O][C][3][3]; GEMM k = tap*C + c
__global__ void k_prep_conv(const float* __restrict__ w, _Float16* __restrict__ dst, int C){
  int Ktot = C*9;
  int nkb = Ktot >> 5;
  int total = nkb * 16 * 512;
  for(int i = blockIdx.x*blockDim.x + threadIdx.x; i < total; i += gridDim.x*blockDim.x){
    int j = i & 15, lane = (i >> 4) & 31, nt = (i >> 9) & 15, kb = i >> 13;
    int k = kb*32 + ((lane >> 4) << 3) + (j & 7) + ((j >> 3) << 4);
    int c = k % C, t = k / C;
    int o = nt*16 + (lane & 15);
    dst[i] = (_Float16)w[(o*C + c)*9 + t];
  }
}

// ---------------------------------------------------------------------------
// Patch MLP: 3 fused pointwise layers (256->256)x3 via WMMA, then 2-logit
// softmax. 32 pixel rows per block, double-buffered activations in LDS.
// Layers are manually unrolled (direct pointers) so weight loads stay on the
// global (SADDR) path instead of flat 64-bit VGPR addressing.
// ---------------------------------------------------------------------------
static __device__ __forceinline__ void mlp_layer(const _Float16* __restrict__ F,
    const float* __restrict__ bias, const _Float16* src, _Float16* dst,
    int mt, int nt0, int l16, int hiL, int lane){
  v8f zero = {};
  v8f acc[4] = {zero, zero, zero, zero};
  for(int kb = 0; kb < 8; ++kb){
    const int r = mt*16 + l16;
    const int k0 = kb*32 + 8*hiL;                 // ISA A 16x32: runs K=[8hi..] and K=[16+8hi..]
    v8h lo = *(const v8h*)(src + r*256 + k0);
    v8h hi = *(const v8h*)(src + r*256 + k0 + 16);
    v16h a = cat16(lo, hi);
    if(kb + 1 < 8) __builtin_prefetch(F + ((kb+1)*16 + nt0)*512, 0, 3);
#pragma unroll
    for(int j = 0; j < 4; ++j){
      v16h bfr = *(const v16h*)(F + (kb*16 + nt0 + j)*512 + lane*16);
      acc[j] = wmma_f16(a, bfr, acc[j]);
    }
  }
#pragma unroll
  for(int j = 0; j < 4; ++j){
    const int n = (nt0 + j)*16 + l16;
    const float bb = bias[n];
#pragma unroll
    for(int rr = 0; rr < 8; ++rr){
      const int m = mt*16 + rr + 8*hiL;           // D layout: vgpr rr -> M=rr (+8 for hi lanes)
      dst[m*256 + n] = (_Float16)leaky_(acc[j][rr] + bb);
    }
  }
  __syncthreads();
}

__global__ __launch_bounds__(256) void k_patch_mlp(
    const float* __restrict__ feat_down,
    const _Float16* __restrict__ f1, const float* __restrict__ b1,
    const _Float16* __restrict__ f2, const float* __restrict__ b2,
    const _Float16* __restrict__ f3, const float* __restrict__ b3,
    const float* __restrict__ pfc_w, const float* __restrict__ pfc_b,
    float* __restrict__ ps){
  __shared__ __align__(16) _Float16 sA[32*256];
  __shared__ __align__(16) _Float16 sB[32*256];
  const int tid = threadIdx.x;
  const int row0 = blockIdx.x * 32;
  for(int i = tid; i < 32*256; i += 256){
    int r = i >> 8, c = i & 255, p = row0 + r, b = p >> 10, pos = p & 1023;
    sA[i] = (_Float16)feat_down[(b*256 + c)*1024 + pos];
  }
  __syncthreads();
  const int wave = tid >> 5, lane = tid & 31, l16 = lane & 15, hiL = lane >> 4;
  const int mt = wave & 1, nt0 = (wave >> 1) * 4;
  mlp_layer(f1, b1, sA, sB, mt, nt0, l16, hiL, lane);
  mlp_layer(f2, b2, sB, sA, mt, nt0, l16, hiL, lane);
  mlp_layer(f3, b3, sA, sB, mt, nt0, l16, hiL, lane);
  for(int r = tid; r < 32; r += 256){
    float l0 = pfc_b[0], l1 = pfc_b[1];
    for(int c = 0; c < 256; ++c){
      float v = (float)sB[r*256 + c];
      l0 += v * pfc_w[c];
      l1 += v * pfc_w[256 + c];
    }
    float mm = fmaxf(l0, l1);
    float e0 = __expf(l0 - mm), e1 = __expf(l1 - mm);
    ps[row0 + r] = e1 / (e0 + e1);
  }
}

// ---------------------------------------------------------------------------
// Gate: bilinear-upsample patch score 32x32 -> 512x512, bad-pixel mean, bm.
// ---------------------------------------------------------------------------
static __device__ __forceinline__ float ps_bilinear(const float* __restrict__ ps, int b, int y, int x){
  float sy = ((float)y + 0.5f) * (1.f/16.f) - 0.5f;
  float sx = ((float)x + 0.5f) * (1.f/16.f) - 0.5f;
  int y0 = (int)floorf(sy), x0 = (int)floorf(sx);
  float wy = sy - (float)y0, wx = sx - (float)x0;
  int y0c = min(max(y0, 0), 31), y1c = min(max(y0 + 1, 0), 31);
  int x0c = min(max(x0, 0), 31), x1c = min(max(x0 + 1, 0), 31);
  const float* p = ps + b*1024;
  float v00 = p[y0c*32 + x0c], v01 = p[y0c*32 + x1c];
  float v10 = p[y1c*32 + x0c], v11 = p[y1c*32 + x1c];
  return v00*(1.f-wy)*(1.f-wx) + v01*(1.f-wy)*wx + v10*wy*(1.f-wx) + v11*wy*wx;
}

__global__ void k_zero_stats(unsigned* u){ if(threadIdx.x < 8) u[threadIdx.x] = 0u; }

__global__ __launch_bounds__(256) void k_gate_stats(const float* __restrict__ xhat,
    const float* __restrict__ ps, float* badsum, unsigned* badcnt){
  __shared__ float ssum[256]; __shared__ unsigned scnt[256];
  int i = blockIdx.x*256 + threadIdx.x;
  float v = 0.f; unsigned c = 0u;
  if(i < B_*H_*W_){
    int b = i >> 18, yx = i & 262143, y = yx >> 9, x = yx & 511;
    float p = ps_bilinear(ps, b, y, x);
    if(p < 0.5f){ v = xhat[i]; c = 1u; }
  }
  ssum[threadIdx.x] = v; scnt[threadIdx.x] = c;
  __syncthreads();
  for(int s = 128; s > 0; s >>= 1){
    if(threadIdx.x < s){ ssum[threadIdx.x] += ssum[threadIdx.x+s]; scnt[threadIdx.x] += scnt[threadIdx.x+s]; }
    __syncthreads();
  }
  if(threadIdx.x == 0){ atomicAdd(badsum, ssum[0]); atomicAdd(badcnt, scnt[0]); }
}

__global__ __launch_bounds__(256) void k_gate_apply(const float* __restrict__ xhat,
    const float* __restrict__ ps, const float* __restrict__ badsum,
    const unsigned* __restrict__ badcnt, float* __restrict__ bm){
  int i = blockIdx.x*256 + threadIdx.x;
  if(i >= B_*H_*W_) return;
  int b = i >> 18, yx = i & 262143, y = yx >> 9, x = yx & 511;
  float p = ps_bilinear(ps, b, y, x);
  float mean = badsum[0] / fmaxf((float)badcnt[0], 1.f);
  bm[i] = (p < 0.5f) ? mean : xhat[i];
}

// ---------------------------------------------------------------------------
// NMS: 7x7 local maxima above threshold, LDS-tiled (16x16 tile + 3 halo).
// ---------------------------------------------------------------------------
__global__ __launch_bounds__(256) void k_nms(const float* __restrict__ bm,
    float* __restrict__ pk_s, int* __restrict__ pk_i, unsigned* __restrict__ cnt){
  __shared__ float t[22*22];
  const int ox = blockIdx.x*16, oy = blockIdx.y*16, b = blockIdx.z;
  const float* img = bm + (size_t)b*H_*W_;
  for(int i = threadIdx.x; i < 22*22; i += 256){
    int ly = i / 22, lx = i % 22;
    int gy = oy + ly - 3, gx = ox + lx - 3;
    float v = -3.0e38f;
    if((unsigned)gy < (unsigned)H_ && (unsigned)gx < (unsigned)W_) v = img[gy*W_ + gx];
    t[i] = v;
  }
  __syncthreads();
  int ly = threadIdx.x >> 4, lx = threadIdx.x & 15;
  int gy = oy + ly, gx = ox + lx;
  float c = t[(ly+3)*22 + (lx+3)];
  if(!(c > NMS_THR_)) return;
  float mx = -3.0e38f;
#pragma unroll
  for(int dy = 0; dy < 7; ++dy)
#pragma unroll
    for(int dx = 0; dx < 7; ++dx)
      mx = fmaxf(mx, t[(ly+dy)*22 + (lx+dx)]);
  if(c >= mx){
    unsigned pos = atomicAdd(&cnt[b], 1u);
    if(pos < CAP_){ pk_s[b*CAP_ + pos] = c; pk_i[b*CAP_ + pos] = gy*W_ + gx; }
  }
}

// ---------------------------------------------------------------------------
// Top-512 per image by iterative block-wide argmax over the peak list.
// ---------------------------------------------------------------------------
__global__ __launch_bounds__(256) void k_topk(float* __restrict__ pk_s,
    const int* __restrict__ pk_i, const unsigned* __restrict__ cnt,
    float* __restrict__ out, int* __restrict__ coord_i){
  __shared__ float ss[256]; __shared__ int si[256];
  const int b = blockIdx.x, tid = threadIdx.x;
  int n = (int)cnt[b]; if(n > CAP_) n = CAP_;
  float* S = pk_s + b*CAP_;
  const int* I = pk_i + b*CAP_;
  for(int r = 0; r < KTOP; ++r){
    float best = -3.0e38f; int bi = -1;
    for(int i = tid; i < n; i += 256){
      float s = S[i];
      if(s > best){ best = s; bi = i; }
    }
    ss[tid] = best; si[tid] = bi;
    __syncthreads();
    for(int s = 128; s > 0; s >>= 1){
      if(tid < s && ss[tid+s] > ss[tid]){ ss[tid] = ss[tid+s]; si[tid] = si[tid+s]; }
      __syncthreads();
    }
    if(tid == 0){
      int sel = si[0];
      float sc = (sel >= 0) ? ss[0] : -__builtin_inff();
      int idx = (sel >= 0) ? I[sel] : 0;
      if(sel >= 0) S[sel] = -3.0e38f;
      int y = idx / W_, x = idx % W_;
      const int g = b*KTOP + r;
      out[8192 + g*2 + 0] = (float)x;          // coords (x,y)
      out[8192 + g*2 + 1] = (float)y;
      out[12288 + g] = sc;                     // top_s
      out[14336 + g] = (sc > NMS_THR_) ? 1.f : 0.f; // valid
      coord_i[g*2 + 0] = x; coord_i[g*2 + 1] = y;
    }
    __syncthreads();
  }
}

// ---------------------------------------------------------------------------
// ROI align: 49 bilinear samples x 64 channels per roi -> f16 pooled.
// ---------------------------------------------------------------------------
__global__ __launch_bounds__(256) void k_roialign(const float* __restrict__ feat_up,
    const int* __restrict__ coord_i, _Float16* __restrict__ pooled){
  const int roi = blockIdx.x, b = roi >> 9;
  const int cx = coord_i[roi*2 + 0], cy = coord_i[roi*2 + 1];
  for(int i = threadIdx.x; i < 49*64; i += 256){
    const int pos = i >> 6, c = i & 63, py = pos / 7, px = pos % 7;
    const float fx = (float)cx - 5.f + ((float)px + 0.5f) * (10.f/7.f);
    const float fy = (float)cy - 5.f + ((float)py + 0.5f) * (10.f/7.f);
    int x0 = (int)floorf(fx), y0 = (int)floorf(fy);
    const float wx = fx - (float)x0, wy = fy - (float)y0;
    const int x0c = min(max(x0, 0), W_-1), x1c = min(max(x0+1, 0), W_-1);
    const int y0c = min(max(y0, 0), H_-1), y1c = min(max(y0+1, 0), H_-1);
    const float* f = feat_up + ((size_t)(b*64 + c))*H_*W_;
    float v = f[y0c*W_ + x0c]*(1.f-wy)*(1.f-wx) + f[y0c*W_ + x1c]*(1.f-wy)*wx
            + f[y1c*W_ + x0c]*wy*(1.f-wx)       + f[y1c*W_ + x1c]*wy*wx;
    pooled[(size_t)roi*49*64 + i] = (_Float16)v;
  }
}

// ---------------------------------------------------------------------------
// Conv layer 1: 64 -> 256, 3x3 SAME on 7x7, implicit GEMM via WMMA.
// One ROI per block. M = 49 positions (padded to 64), N = 256, K = 576.
// Input tile staged via TDM; output drained via TDM.
// ---------------------------------------------------------------------------
__global__ __launch_bounds__(256) void k_conv1(const _Float16* __restrict__ in,
    const _Float16* __restrict__ Wf, const float* __restrict__ bias,
    _Float16* __restrict__ out){
  __shared__ __align__(16) _Float16 sin_[49*64];
  __shared__ __align__(16) _Float16 sout_[49*256];
  const int roi = blockIdx.x, tid = threadIdx.x;
  stage_in(sin_, in + (size_t)roi*49*64, 49*64, tid);
  const int wave = tid >> 5, lane = tid & 31, l16 = lane & 15, hiL = lane >> 4;
  const int nt0 = wave * 2;
  v8f zero = {};
  v8f acc[4][2];
#pragma unroll
  for(int mt = 0; mt < 4; ++mt){ acc[mt][0] = zero; acc[mt][1] = zero; }
  for(int kb = 0; kb < 18; ++kb){
    const int t = kb >> 1, cb = (kb & 1) * 32;   // k = tap*64 + c; 32-block stays in one tap
    const int dy = t/3 - 1, dx = t%3 - 1;
    v16h afr[4];
#pragma unroll
    for(int mt = 0; mt < 4; ++mt){
      const int m = mt*16 + l16;
      v8h lo = {}, hi = {};
      if(m < 49){
        const int py = m / 7, px = m % 7;
        const int iy = py + dy, ix = px + dx;
        if((unsigned)iy < 7u && (unsigned)ix < 7u){
          const int c0 = cb + 8*hiL;
          const _Float16* p = sin_ + (iy*7 + ix)*64 + c0;
          lo = *(const v8h*)p;
          hi = *(const v8h*)(p + 16);
        }
      }
      afr[mt] = cat16(lo, hi);
    }
    if(kb + 1 < 18) __builtin_prefetch(Wf + ((kb+1)*16 + nt0)*512, 0, 3);
#pragma unroll
    for(int j = 0; j < 2; ++j){
      v16h bfr = *(const v16h*)(Wf + (kb*16 + nt0 + j)*512 + lane*16);
#pragma unroll
      for(int mt = 0; mt < 4; ++mt) acc[mt][j] = wmma_f16(afr[mt], bfr, acc[mt][j]);
    }
  }
#pragma unroll
  for(int j = 0; j < 2; ++j){
    const int n = (nt0 + j)*16 + l16;
    const float bb = bias[n];
#pragma unroll
    for(int mt = 0; mt < 4; ++mt)
#pragma unroll
      for(int rr = 0; rr < 8; ++rr){
        const int m = mt*16 + rr + 8*hiL;
        if(m < 49) sout_[m*256 + n] = (_Float16)leaky_(acc[mt][j][rr] + bb);
      }
  }
  __syncthreads();
  drain_out(out + (size_t)roi*49*256, sout_, 49*256, tid);
}

// ---------------------------------------------------------------------------
// Conv layers 2/3 core: 256 -> 256. K = 2304 (72 WMMA K-steps / wave tile).
// ---------------------------------------------------------------------------
static __device__ __forceinline__ void conv256_body(const _Float16* __restrict__ in,
    const _Float16* __restrict__ Wf, const float* __restrict__ bias,
    _Float16* sin_, _Float16* sout_, int roi, int tid){
  stage_in(sin_, in + (size_t)roi*49*256, 49*256, tid);
  const int wave = tid >> 5, lane = tid & 31, l16 = lane & 15, hiL = lane >> 4;
  const int nt0 = wave * 2;
  v8f zero = {};
  v8f acc[4][2];
#pragma unroll
  for(int mt = 0; mt < 4; ++mt){ acc[mt][0] = zero; acc[mt][1] = zero; }
  for(int kb = 0; kb < 72; ++kb){
    const int t = kb >> 3, cb = (kb & 7) * 32;   // k = tap*256 + c
    const int dy = t/3 - 1, dx = t%3 - 1;
    v16h afr[4];
#pragma unroll
    for(int mt = 0; mt < 4; ++mt){
      const int m = mt*16 + l16;
      v8h lo = {}, hi = {};
      if(m < 49){
        const int py = m / 7, px = m % 7;
        const int iy = py + dy, ix = px + dx;
        if((unsigned)iy < 7u && (unsigned)ix < 7u){
          const int c0 = cb + 8*hiL;
          const _Float16* p = sin_ + (iy*7 + ix)*256 + c0;
          lo = *(const v8h*)p;
          hi = *(const v8h*)(p + 16);
        }
      }
      afr[mt] = cat16(lo, hi);
    }
    if(kb + 1 < 72) __builtin_prefetch(Wf + ((kb+1)*16 + nt0)*512, 0, 3);
#pragma unroll
    for(int j = 0; j < 2; ++j){
      v16h bfr = *(const v16h*)(Wf + (kb*16 + nt0 + j)*512 + lane*16);
#pragma unroll
      for(int mt = 0; mt < 4; ++mt) acc[mt][j] = wmma_f16(afr[mt], bfr, acc[mt][j]);
    }
  }
#pragma unroll
  for(int j = 0; j < 2; ++j){
    const int n = (nt0 + j)*16 + l16;
    const float bb = bias[n];
#pragma unroll
    for(int mt = 0; mt < 4; ++mt)
#pragma unroll
      for(int rr = 0; rr < 8; ++rr){
        const int m = mt*16 + rr + 8*hiL;
        if(m < 49) sout_[m*256 + n] = (_Float16)leaky_(acc[mt][j][rr] + bb);
      }
  }
  __syncthreads();
}

__global__ __launch_bounds__(256) void k_conv_mid(const _Float16* __restrict__ in,
    const _Float16* __restrict__ Wf, const float* __restrict__ bias,
    _Float16* __restrict__ out){
  __shared__ __align__(16) _Float16 sin_[49*256];
  __shared__ __align__(16) _Float16 sout_[49*256];
  const int roi = blockIdx.x, tid = threadIdx.x;
  conv256_body(in, Wf, bias, sin_, sout_, roi, tid);
  drain_out(out + (size_t)roi*49*256, sout_, 49*256, tid);
}

// Last conv layer fuses AdaptiveMaxPool(1) + FC(256->4).
__global__ __launch_bounds__(256) void k_conv_last(const _Float16* __restrict__ in,
    const _Float16* __restrict__ Wf, const float* __restrict__ bias,
    const float* __restrict__ qfcw, const float* __restrict__ qfcb,
    float* __restrict__ outL){
  __shared__ __align__(16) _Float16 sin_[49*256];
  __shared__ __align__(16) _Float16 sout_[49*256];
  __shared__ float sh[256];
  __shared__ float red[256];
  const int roi = blockIdx.x, tid = threadIdx.x;
  conv256_body(in, Wf, bias, sin_, sout_, roi, tid);
  for(int c = tid; c < 256; c += 256){
    float mx = -3.0e38f;
    for(int m = 0; m < 49; ++m) mx = fmaxf(mx, (float)sout_[m*256 + c]);
    sh[c] = mx;
  }
  __syncthreads();
  const int cls = tid >> 6, jj = tid & 63;
  float part = 0.f;
  for(int c = jj; c < 256; c += 64) part += sh[c] * qfcw[cls*256 + c];
  red[tid] = part;
  __syncthreads();
  for(int s = 32; s > 0; s >>= 1){
    if(jj < s) red[tid] += red[tid + s];
    __syncthreads();
  }
  if(jj == 0) outL[roi*4 + cls] = red[tid] + qfcb[cls];
}

// ---------------------------------------------------------------------------
extern "C" void kernel_launch(void* const* d_in, const int* in_sizes, int n_in,
                              void* d_out, int out_size, void* d_ws, size_t ws_size,
                              hipStream_t stream) {
  const float* xhat      = (const float*)d_in[0];
  const float* feat_down = (const float*)d_in[1];
  const float* feat_up   = (const float*)d_in[2];
  const float* pw1 = (const float*)d_in[3];  const float* pb1 = (const float*)d_in[4];
  const float* pw2 = (const float*)d_in[5];  const float* pb2 = (const float*)d_in[6];
  const float* pw3 = (const float*)d_in[7];  const float* pb3 = (const float*)d_in[8];
  const float* pfc_w = (const float*)d_in[9]; const float* pfc_b = (const float*)d_in[10];
  const float* qw1 = (const float*)d_in[11]; const float* qb1 = (const float*)d_in[12];
  const float* qw2 = (const float*)d_in[13]; const float* qb2 = (const float*)d_in[14];
  const float* qw3 = (const float*)d_in[15]; const float* qb3 = (const float*)d_in[16];
  const float* qfc_w = (const float*)d_in[17]; const float* qfc_b = (const float*)d_in[18];

  char* ws = (char*)d_ws;
  size_t off = 0;
  auto alloc = [&](size_t bytes) -> char* {
    char* p = ws + off; off = (off + bytes + 255) & ~(size_t)255; return p;
  };
  float*    ps      = (float*)alloc(4096u * 4);
  unsigned* stats   = (unsigned*)alloc(256);                      // [0]=badsum(f32) [1]=badcnt [2..5]=peak cnt
  float*    bm      = (float*)alloc((size_t)B_*H_*W_*4);
  float*    pk_s    = (float*)alloc((size_t)B_*CAP_*4);
  int*      pk_i    = (int*)alloc((size_t)B_*CAP_*4);
  int*      coord_i = (int*)alloc((size_t)B_*KTOP*2*4);
  _Float16* pooled  = (_Float16*)alloc((size_t)2048*49*64*2);
  _Float16* acta    = (_Float16*)alloc((size_t)2048*49*256*2);
  _Float16* actb    = (_Float16*)alloc((size_t)2048*49*256*2);
  _Float16* fp1 = (_Float16*)alloc((size_t)8*16*512*2);
  _Float16* fp2 = (_Float16*)alloc((size_t)8*16*512*2);
  _Float16* fp3 = (_Float16*)alloc((size_t)8*16*512*2);
  _Float16* fq1 = (_Float16*)alloc((size_t)18*16*512*2);
  _Float16* fq2 = (_Float16*)alloc((size_t)72*16*512*2);
  _Float16* fq3 = (_Float16*)alloc((size_t)72*16*512*2);
  float* out = (float*)d_out;

  // weight packing into WMMA B-fragment order
  k_prep_dense<<<64, 256, 0, stream>>>(pw1, fp1, 256);
  k_prep_dense<<<64, 256, 0, stream>>>(pw2, fp2, 256);
  k_prep_dense<<<64, 256, 0, stream>>>(pw3, fp3, 256);
  k_prep_conv <<<144, 256, 0, stream>>>(qw1, fq1, 64);
  k_prep_conv <<<576, 256, 0, stream>>>(qw2, fq2, 256);
  k_prep_conv <<<576, 256, 0, stream>>>(qw3, fq3, 256);

  k_zero_stats<<<1, 32, 0, stream>>>(stats);
  k_patch_mlp<<<128, 256, 0, stream>>>(feat_down, fp1, pb1, fp2, pb2, fp3, pb3, pfc_w, pfc_b, ps);
  k_gate_stats<<<4096, 256, 0, stream>>>(xhat, ps, (float*)stats, stats + 1);
  k_gate_apply<<<4096, 256, 0, stream>>>(xhat, ps, (const float*)stats, stats + 1, bm);

  dim3 ng(32, 32, 4);
  k_nms<<<ng, 256, 0, stream>>>(bm, pk_s, pk_i, stats + 2);
  k_topk<<<4, 256, 0, stream>>>(pk_s, pk_i, stats + 2, out, coord_i);

  k_roialign<<<2048, 256, 0, stream>>>(feat_up, coord_i, pooled);
  k_conv1   <<<2048, 256, 0, stream>>>(pooled, fq1, qb1, acta);
  k_conv_mid<<<2048, 256, 0, stream>>>(acta, fq2, qb2, actb);
  k_conv_last<<<2048, 256, 0, stream>>>(actb, fq3, qb3, qfc_w, qfc_b, out);
}
